// IndexingIssues_39539468927803
// MI455X (gfx1250) — matrix operand order repaired and checked by reading, not compiled, not run on hardware
//
#include <hip/hip_runtime.h>
#include <cstdint>

typedef __attribute__((ext_vector_type(16))) __bf16 v16bf;
typedef __attribute__((ext_vector_type(8)))  __bf16 v8bf;
typedef __attribute__((ext_vector_type(8)))  float  v8f;
typedef __attribute__((ext_vector_type(4)))  float  v4f;
typedef __attribute__((ext_vector_type(4)))  int    v4i;

#define S_LEN 256
#define BATCH 128
#define IN    1024
#define HID   512
#define G4    (4 * HID)          // 2048 gate columns
#define SB    (S_LEN * BATCH)    // 32768 rows of the big x-GEMM
#define KC    128                // K-chunk staged in LDS per buffer

// ---------------- CDNA5 async global->LDS path ----------------
// Builtin signature (decoded from clang diagnostics):
//   void __builtin_amdgcn_global_load_async_to_lds_b128(
//       v4i addrspace(1)* src_global, v4i addrspace(3)* dst_lds,
//       imm int offset, imm int cpol)

#if __has_builtin(__builtin_amdgcn_global_load_async_to_lds_b128)
#warning "CDNA5 probe: __builtin_amdgcn_global_load_async_to_lds_b128 AVAILABLE"
#define HAVE_ASYNC_LDS 1
#else
#warning "CDNA5 probe: __builtin_amdgcn_global_load_async_to_lds_b128 MISSING (sync fallback)"
#define HAVE_ASYNC_LDS 0
#endif

typedef __attribute__((address_space(1))) v4i glob_v4i;
typedef __attribute__((address_space(3))) v4i lds_v4i;

__device__ __forceinline__ void async_copy_b128(__bf16* dst_lds, const __bf16* src_glob) {
#if HAVE_ASYNC_LDS
    __builtin_amdgcn_global_load_async_to_lds_b128(
        (glob_v4i*)(uintptr_t)src_glob,
        (lds_v4i*)(uintptr_t)dst_lds,
        /*offset=*/0, /*cpol=*/0);
#else
    *(v8bf*)dst_lds = *(const v8bf*)src_glob;   // synchronous fallback
#endif
}

__device__ __forceinline__ void wait_async_zero() {
#if __has_builtin(__builtin_amdgcn_s_wait_asynccnt)
    __builtin_amdgcn_s_wait_asynccnt(0);
#else
    asm volatile("s_wait_asynccnt 0x0" ::: "memory");
#endif
}

// ---------------- WMMA fragment helpers (contiguous k order k = 16*hi + s, same for A and B) ---

// A fragment: 16x32 bf16 tile, row-major, leading dim ld. Lane holds row
// M=lane&15; contiguous K run hi*16..+15 (two b128 loads).
__device__ __forceinline__ v16bf load_afrag(const __bf16* tile, int ld) {
    int lane = threadIdx.x & 31;
    int r    = lane & 15;
    int hi   = lane >> 4;
    const __bf16* base = tile + (size_t)r * ld + hi * 16;
    union { v16bf v; v8bf h[2]; } u;
    u.h[0] = *(const v8bf*)(base);
    u.h[1] = *(const v8bf*)(base + 8);
    return u.v;
}

// B fragment from global: B[k][n] = W[n][k], W row-major, leading dim ld.
// Lane holds column n=lane&15, contiguous K run hi*16..+15 (32B contiguous).
__device__ __forceinline__ v16bf load_bfrag(const __bf16* wtile, int ld) {
    int lane = threadIdx.x & 31;
    int n    = lane & 15;
    int hi   = lane >> 4;
    const v8bf* p = (const v8bf*)(wtile + (size_t)n * ld + hi * 16);
    union { v16bf v; v8bf h[2]; } u;
    u.h[0] = p[0];
    u.h[1] = p[1];
    return u.v;
}

// B fragment from an LDS stage buffer laid out [64 cols][KC k-elems].
__device__ __forceinline__ v16bf load_bfrag_lds(const __bf16 (*stage)[KC], int j, int kk) {
    int lane = threadIdx.x & 31;
    int n    = lane & 15;
    int hi   = lane >> 4;
    const __bf16* p = &stage[j * 16 + n][kk * 32 + hi * 16];
    union { v16bf v; v8bf h[2]; } u;
    u.h[0] = ((const v8bf*)p)[0];
    u.h[1] = ((const v8bf*)p)[1];
    return u.v;
}

__device__ __forceinline__ v8f wmma_bf16(v16bf a, v16bf b, v8f c) {
    return __builtin_amdgcn_wmma_f32_16x16x32_bf16(
        false, a, false, b, (short)0, c, false, false);
}

__device__ __forceinline__ float sigmoid_f(float x) {
    return 1.0f / (1.0f + __expf(-x));
}

// ---------------- Utility kernels ----------------

// 8 elements per thread: 2x b128 f32 loads -> 1x b128 bf16 store.
__global__ void cvt_f32_bf16_v8(const float* __restrict__ in, __bf16* __restrict__ out, int n8) {
    int i = blockIdx.x * blockDim.x + threadIdx.x;
    if (i < n8) {
        v4f x0 = ((const v4f*)in)[i * 2 + 0];
        v4f x1 = ((const v4f*)in)[i * 2 + 1];
        v8bf o;
#pragma unroll
        for (int t = 0; t < 4; ++t) {
            o[t]     = (__bf16)x0[t];
            o[4 + t] = (__bf16)x1[t];
        }
        ((v8bf*)out)[i] = o;
    }
}

__global__ void init_state(float* __restrict__ c, __bf16* __restrict__ h0) {
    int i = blockIdx.x * blockDim.x + threadIdx.x;
    if (i < BATCH * HID) {
        c[i]  = 0.0f;
        h0[i] = (__bf16)0.0f;
    }
}

// ---------------- Big parallel GEMM: G = X @ W_ih^T + (b_ih + b_hh) ----------------
// Block = 8 waves covering 128 rows x 64 cols. The 64-column W_ih tile (shared
// by all 8 waves) is staged global->LDS with async copies, double-buffered in
// KC=128 K-chunks (2 x 16 KB LDS). A-fragments are prefetched into registers
// for a whole chunk before the 16 WMMAs, so load waits overlap matrix ops.
__global__ __launch_bounds__(256) void gemm_xw(
    const __bf16* __restrict__ xb,    // [SB, IN] bf16
    const __bf16* __restrict__ wih,   // [G4, IN] bf16 (row j = gate col j)
    const float*  __restrict__ b_ih,  // [G4]
    const float*  __restrict__ b_hh,  // [G4]
    float*        __restrict__ G)     // [SB, G4] f32
{
    __shared__ __bf16 bstage[2][64][KC];   // 32 KB

    int tid  = threadIdx.x;
    int wave = tid >> 5;
    int lane = tid & 31;
    int mblk = blockIdx.x >> 5;    // 0..255 : 128-row block
    int ng   = blockIdx.x & 31;    // 0..31  : 64-col group
    int m0   = mblk * 128 + wave * 16;
    int n0   = ng * 64;
    int ncol = lane & 15;
    int hi   = lane >> 4;

    // Cooperative stage of W_ih[n0..n0+63][kc..kc+KC) into bstage[buf].
    // 64 rows x 256 B = 1024 b128 transfers, 4 per thread.
    auto stage = [&](int kc, int buf) {
#pragma unroll
        for (int i = 0; i < 4; ++i) {
            int idx = tid + i * 256;      // 0..1023
            int row = idx >> 4;           // 0..63
            int seg = idx & 15;           // 0..15 (16B segments)
            async_copy_b128(&bstage[buf][row][seg * 8],
                            wih + (size_t)(n0 + row) * IN + kc + seg * 8);
        }
    };

    v8f acc[4];
#pragma unroll
    for (int j = 0; j < 4; ++j) {
        float bv = b_ih[n0 + j * 16 + ncol] + b_hh[n0 + j * 16 + ncol];
#pragma unroll
        for (int v = 0; v < 8; ++v) acc[j][v] = bv;
    }

    const __bf16* arow = xb + (size_t)m0 * IN;
    const int NCHUNK = IN / KC;   // 8

    stage(0, 0);
    wait_async_zero();
    __syncthreads();

    for (int c = 0; c < NCHUNK; ++c) {
        int buf = c & 1;
        if (c + 1 < NCHUNK) stage((c + 1) * KC, buf ^ 1);   // overlap next stage

        // Prefetch all A fragments of this chunk (4 x 8 VGPRs).
        v16bf a[KC / 32];
#pragma unroll
        for (int kk = 0; kk < KC / 32; ++kk)
            a[kk] = load_afrag(arow + c * KC + kk * 32, IN);

#pragma unroll
        for (int kk = 0; kk < KC / 32; ++kk) {
#pragma unroll
            for (int j = 0; j < 4; ++j)
                acc[j] = wmma_bf16(a[kk], load_bfrag_lds(bstage[buf], j, kk), acc[j]);
        }

        wait_async_zero();   // next buffer landed
        __syncthreads();     // everyone done reading current buffer
    }

#pragma unroll
    for (int j = 0; j < 4; ++j)
#pragma unroll
        for (int v = 0; v < 8; ++v)
            G[(size_t)(m0 + v + 8 * hi) * G4 + (n0 + j * 16 + ncol)] = acc[j][v];
}

// ---------------- Fused recurrent step ----------------
// gates = G[s] + h_in @ W_hh^T ; 4 waves = i/f/g/o tiles, meet in LDS, update c/h.
// K-loop is depth-1 register-pipelined: next A/B fragments are in flight while
// the current WMMA executes (critical-path latency, 256 sequential launches).
__global__ __launch_bounds__(128) void lstm_step(
    const __bf16* __restrict__ h_in,  // [BATCH, HID] bf16
    const __bf16* __restrict__ whh,   // [G4, HID] bf16
    const float*  __restrict__ G,     // [SB, G4] f32 (x-part + bias)
    float*        __restrict__ c,     // [BATCH, HID] f32 (in-place)
    __bf16*       __restrict__ h_out, // [BATCH, HID] bf16
    int s)
{
    __shared__ float lds[4][16][16];

    int wv   = threadIdx.x >> 5;     // gate 0..3
    int lane = threadIdx.x & 31;
    int bt   = blockIdx.x >> 5;      // batch tile 0..7
    int ht   = blockIdx.x & 31;      // hidden tile 0..31
    int m0   = bt << 4;
    int n0   = wv * HID + (ht << 4);
    int ncol = lane & 15;
    int hi   = lane >> 4;

    const float* Grow = G + (size_t)(s * BATCH + m0) * G4 + n0 + ncol;
    v8f acc;
#pragma unroll
    for (int v = 0; v < 8; ++v) acc[v] = Grow[(size_t)(v + 8 * hi) * G4];

    const __bf16* arow = h_in + (size_t)m0 * HID;
    const __bf16* wrow = whh + (size_t)n0 * HID;

    v16bf a0 = load_afrag(arow, HID);
    v16bf b0 = load_bfrag(wrow, HID);
#pragma unroll
    for (int k = 0; k < HID; k += 32) {
        int kn = (k + 32 < HID) ? (k + 32) : k;   // last iter: harmless reload
        v16bf a1 = load_afrag(arow + kn, HID);
        v16bf b1 = load_bfrag(wrow + kn, HID);
        acc = wmma_bf16(a0, b0, acc);
        a0 = a1;
        b0 = b1;
    }

#pragma unroll
    for (int v = 0; v < 8; ++v) lds[wv][v + 8 * hi][ncol] = acc[v];
    __syncthreads();

    for (int idx = threadIdx.x; idx < 256; idx += 128) {
        int m = idx >> 4;
        int n = idx & 15;
        float iv = lds[0][m][n];
        float fv = lds[1][m][n];
        float gv = lds[2][m][n];
        float ov = lds[3][m][n];
        size_t p = (size_t)(m0 + m) * HID + (ht << 4) + n;
        float co = c[p];
        float cn = sigmoid_f(fv) * co + sigmoid_f(iv) * tanhf(gv);
        c[p]     = cn;
        h_out[p] = (__bf16)(sigmoid_f(ov) * tanhf(cn));
    }
}

// ---------------- Deterministic single-block reduction of final h ----------------
__global__ __launch_bounds__(256) void reduce_h(const __bf16* __restrict__ h,
                                                float* __restrict__ out) {
    __shared__ float sm[256];
    float s = 0.0f;
    for (int i = threadIdx.x; i < BATCH * HID; i += 256) s += (float)h[i];
    sm[threadIdx.x] = s;
    __syncthreads();
    for (int o = 128; o > 0; o >>= 1) {
        if ((int)threadIdx.x < o) sm[threadIdx.x] += sm[threadIdx.x + o];
        __syncthreads();
    }
    if (threadIdx.x == 0) out[0] = sm[0];
}

// ---------------- Host launcher ----------------
extern "C" void kernel_launch(void* const* d_in, const int* in_sizes, int n_in,
                              void* d_out, int out_size, void* d_ws, size_t ws_size,
                              hipStream_t stream) {
    (void)in_sizes; (void)n_in; (void)out_size; (void)ws_size;

    const float* x    = (const float*)d_in[0];  // [S, B, I]
    const float* w_ih = (const float*)d_in[1];  // [4H, I]
    const float* w_hh = (const float*)d_in[2];  // [4H, H]
    const float* b_ih = (const float*)d_in[3];  // [4H]
    const float* b_hh = (const float*)d_in[4];  // [4H]
    float* out = (float*)d_out;

    char*  ws  = (char*)d_ws;
    size_t off = 0;
    auto carve = [&](size_t bytes) -> void* {
        void* p = ws + off;
        off += (bytes + 255) & ~(size_t)255;
        return p;
    };

    __bf16* xb   = (__bf16*)carve((size_t)SB * IN * sizeof(__bf16));    //  67 MB
    __bf16* wihb = (__bf16*)carve((size_t)G4 * IN * sizeof(__bf16));    //   4 MB
    __bf16* whhb = (__bf16*)carve((size_t)G4 * HID * sizeof(__bf16));   //   2 MB
    float*  G    = (float*) carve((size_t)SB * G4 * sizeof(float));     // 268 MB
    float*  c    = (float*) carve((size_t)BATCH * HID * sizeof(float));
    __bf16* h0   = (__bf16*)carve((size_t)BATCH * HID * sizeof(__bf16));
    __bf16* h1   = (__bf16*)carve((size_t)BATCH * HID * sizeof(__bf16));

    cvt_f32_bf16_v8<<<(SB * IN / 8) / 256, 256, 0, stream>>>(x, xb, SB * IN / 8);
    cvt_f32_bf16_v8<<<(G4 * IN / 8) / 256, 256, 0, stream>>>(w_ih, wihb, G4 * IN / 8);
    cvt_f32_bf16_v8<<<(G4 * HID / 8) / 256, 256, 0, stream>>>(w_hh, whhb, G4 * HID / 8);
    init_state<<<(BATCH * HID) / 256, 256, 0, stream>>>(c, h0);

    // Hoisted input GEMM: all S*B rows in parallel (2/3 of total FLOPs).
    gemm_xw<<<(SB / 128) * (G4 / 64), 256, 0, stream>>>(xb, wihb, b_ih, b_hh, G);

    // Sequential recurrence: one fused kernel per timestep, h double-buffered.
    __bf16* hin = h0;
    __bf16* hout = h1;
    for (int s = 0; s < S_LEN; ++s) {
        lstm_step<<<(BATCH / 16) * (HID / 16), 128, 0, stream>>>(hin, whhb, G, c, hout, s);
        __bf16* t = hin; hin = hout; hout = t;
    }

    reduce_h<<<1, 256, 0, stream>>>(hin, out);
}
